// BGCFLayer_35613868818934
// MI455X (gfx1250) — matrix operand order repaired and checked by reading, not compiled, not run on hardware
//
#include <hip/hip_runtime.h>
#include <math.h>

#define NU 8192
#define NI 8192
#define DD 64
#define BB 1024
#define EPSF 1e-6f

typedef __attribute__((ext_vector_type(16))) __bf16 v16bf;
typedef __attribute__((ext_vector_type(8)))  __bf16 v8bf;
typedef __attribute__((ext_vector_type(8)))  float  v8f;

__device__ __forceinline__ v8f wmma_bf(v16bf a, v16bf b, v8f c) {
  return __builtin_amdgcn_wmma_f32_16x16x32_bf16(false, a, false, b, (short)0, c,
                                                 false, false);
}

// ---- WMMA operand loaders ---------------------------------------------------
// A tile mapping: lane low4 = M row, half = lane>>4, K = 16*half + e
// -> 16 contiguous elements at +16*half.
__device__ __forceinline__ v16bf load_a_bf(const __bf16* base, int half) {
  const __bf16* p = base + half * 16;
  v8bf lo = *(const v8bf*)p;
  v8bf hi = *(const v8bf*)(p + 8);
  v16bf o;
#pragma unroll
  for (int e = 0; e < 8; ++e) { o[e] = lo[e]; o[e + 8] = hi[e]; }
  return o;
}
// B tile mapping: lane low4 = N col, K = 16*half + e -> 16 contiguous elements.
__device__ __forceinline__ v16bf load_b_bf(const __bf16* p) {
  v8bf lo = *(const v8bf*)p;
  v8bf hi = *(const v8bf*)(p + 8);
  v16bf o;
#pragma unroll
  for (int e = 0; e < 8; ++e) { o[e] = lo[e]; o[e + 8] = hi[e]; }
  return o;
}
// A operand from f32 memory (row-major, contiguous): convert + optional sum.
__device__ __forceinline__ v16bf load_a_f32row(const float* p, float* dsum) {
  v16bf o;
  float s = 0.f;
#pragma unroll
  for (int e = 0; e < 8; ++e) { float x = p[e];      s += x; o[e]     = (__bf16)x; }
#pragma unroll
  for (int e = 0; e < 8; ++e) { float x = p[8 + e]; s += x; o[e + 8] = (__bf16)x; }
  if (dsum) *dsum += s;
  return o;
}
// A operand from f32 memory, K-strided (transposed adjacency columns).
// Per element, lanes 0-15 read 16 consecutive floats -> coalesced 64B segments.
__device__ __forceinline__ v16bf load_a_f32col(const float* src, long base, long ks,
                                               int half, float* dsum) {
  v16bf o;
  float s = 0.f;
#pragma unroll
  for (int e = 0; e < 16; ++e) {
    const int k = half * 16 + e;
    const float x = src[base + (long)k * ks];
    s += x;
    o[e] = (__bf16)x;
  }
  if (dsum) *dsum += s;
  return o;
}

// ---------------------------------------------------------------------------
// Prep: fp32 embeddings -> bf16 row-major + bf16 transposed [64][nrows].
// ---------------------------------------------------------------------------
__global__ __launch_bounds__(256)
void cvt_emb_kernel(const float* __restrict__ src, __bf16* __restrict__ row_bf,
                    __bf16* __restrict__ colT_bf, const int nrows) {
  const int i = blockIdx.x * 256 + threadIdx.x;
  if (i >= nrows * DD) return;
  const int r = i >> 6, d = i & 63;
  const __bf16 b = (__bf16)src[i];
  row_bf[i] = b;
  colT_bf[(long)d * nrows + r] = b;
}
__global__ __launch_bounds__(256)
void cvt_wT_kernel(const float* __restrict__ W, __bf16* __restrict__ WT) {
  const int i = blockIdx.x * 256 + threadIdx.x;
  if (i >= DD * DD) return;
  const int k = i >> 6, n = i & 63;
  WT[n * DD + k] = (__bf16)W[i];
}

// ---------------------------------------------------------------------------
// Pass 1: fused attention + plain aggregation for sampled users.
//   a1[b] = (sum_i exp(u.i)*adj[u,i]*item_emb[i]) / (rowsum + eps)
//   a2[b] = (sum_i adj[u,i]*item_emb[i]) / (deg_u + eps)
// One block = 16 gathered users; 8 waves each own a 1024-wide item slice.
// ---------------------------------------------------------------------------
__global__ __launch_bounds__(256)
void attn_user_kernel(const int* __restrict__ users,
                      const __bf16* __restrict__ ue_bf,
                      const __bf16* __restrict__ ie_bf,
                      const __bf16* __restrict__ ieT_bf,
                      const float* __restrict__ adj,
                      float* __restrict__ a1_out, float* __restrict__ a2_out) {
  __shared__ float  smem[8192];       // 32KB: cross-wave acc reduction
  __shared__ __bf16 sS[8 * 512];      // 8KB: exp-score staging (bf16)
  __shared__ __bf16 sA[8 * 512];      // 8KB: adjacency staging (bf16)
  __shared__ float  lds_rs[8][16];
  __shared__ float  lds_dg[8][16];
  __shared__ int    uidx[16];

  const int lane = threadIdx.x & 31;
  const int wv   = threadIdx.x >> 5;
  const int half = lane >> 4;
  const int ln   = lane & 15;
  const int t    = blockIdx.x;

  if (threadIdx.x < 16) uidx[threadIdx.x] = users[t * 16 + threadIdx.x];
  __syncthreads();

  const v16bf Au0 = load_a_bf(ue_bf + (long)uidx[ln] * DD, half);
  const v16bf Au1 = load_a_bf(ue_bf + (long)uidx[ln] * DD + 32, half);

  const v8f vz = {0.f, 0.f, 0.f, 0.f, 0.f, 0.f, 0.f, 0.f};
  v8f acc1[4] = {vz, vz, vz, vz};
  v8f acc2[4] = {vz, vz, vz, vz};
  float rs[8] = {0, 0, 0, 0, 0, 0, 0, 0};
  float dg[8] = {0, 0, 0, 0, 0, 0, 0, 0};

  __bf16* pS = sS + wv * 512;
  __bf16* pA = sA + wv * 512;

  long arow[8];
#pragma unroll
  for (int r = 0; r < 8; ++r) arow[r] = (long)uidx[r + 8 * half] * NI;

  for (int it = 0; it < 32; ++it) {
    const int j0 = wv * 1024 + it * 32;
    // prefetch the adjacency stream a couple of chunks ahead
    __builtin_prefetch(&adj[arow[0] + j0 + 96], 0, 1);
#pragma unroll
    for (int sub = 0; sub < 2; ++sub) {
      const int itm = j0 + sub * 16 + ln;
      const v16bf B0 = load_b_bf(ie_bf + (long)itm * DD + 16 * half);
      const v16bf B1 = load_b_bf(ie_bf + (long)itm * DD + 32 + 16 * half);
      v8f p = wmma_bf(Au0, B0, vz);          // P = U_tile @ I_chunk^T (K=64)
      p = wmma_bf(Au1, B1, p);
#pragma unroll
      for (int r = 0; r < 8; ++r) {          // C layout: M=r+8*half, N=ln
        const int m = r + 8 * half;
        const float av = adj[arow[r] + itm];
        const float s  = __expf(p[r]) * av;
        rs[r] += s;
        dg[r] += av;
        pS[m * 32 + sub * 16 + ln] = (__bf16)s;   // stage to LDS for transpose
        pA[m * 32 + sub * 16 + ln] = (__bf16)av;
      }
    }
    __syncthreads();
    const v16bf As = load_a_bf(pS + ln * 32, half);   // S as A-operand
    const v16bf Aa = load_a_bf(pA + ln * 32, half);   // adj as A-operand
#pragma unroll
    for (int g = 0; g < 4; ++g) {
      const v16bf B2 = load_b_bf(ieT_bf + (long)(g * 16 + ln) * NI + j0 + 16 * half);
      acc1[g] = wmma_bf(As, B2, acc1[g]);    // acc += S @ item_emb_chunk
      acc2[g] = wmma_bf(Aa, B2, acc2[g]);    // acc += adj @ item_emb_chunk
    }
    __syncthreads();
  }

  // reduce rowsum / degree over the 16 lanes of each half-group
#pragma unroll
  for (int r = 0; r < 8; ++r) {
    float v = rs[r], d = dg[r];
#pragma unroll
    for (int off = 1; off < 16; off <<= 1) {
      v += __shfl_xor(v, off, 32);
      d += __shfl_xor(d, off, 32);
    }
    rs[r] = v; dg[r] = d;
  }
  if (ln == 0) {
#pragma unroll
    for (int r = 0; r < 8; ++r) {
      lds_rs[wv][r + 8 * half] = rs[r];
      lds_dg[wv][r + 8 * half] = dg[r];
    }
  }
  __syncthreads();
  // cross-wave reduction of acc1, then acc2 (reuse smem)
#pragma unroll
  for (int g = 0; g < 4; ++g)
#pragma unroll
    for (int r = 0; r < 8; ++r)
      smem[(wv * 16 + r + 8 * half) * 64 + g * 16 + ln] = acc1[g][r];
  __syncthreads();
  for (int i2 = (int)threadIdx.x; i2 < 1024; i2 += 256) {
    const int m = i2 >> 6, d = i2 & 63;
    float s1 = 0.f, R = 0.f;
    for (int w = 0; w < 8; ++w) { s1 += smem[(w * 16 + m) * 64 + d]; R += lds_rs[w][m]; }
    a1_out[(long)(t * 16 + m) * DD + d] = s1 / (R + EPSF);
  }
  __syncthreads();
#pragma unroll
  for (int g = 0; g < 4; ++g)
#pragma unroll
    for (int r = 0; r < 8; ++r)
      smem[(wv * 16 + r + 8 * half) * 64 + g * 16 + ln] = acc2[g][r];
  __syncthreads();
  for (int i2 = (int)threadIdx.x; i2 < 1024; i2 += 256) {
    const int m = i2 >> 6, d = i2 & 63;
    float s2 = 0.f, Dg = 0.f;
    for (int w = 0; w < 8; ++w) { s2 += smem[(w * 16 + m) * 64 + d]; Dg += lds_dg[w][m]; }
    a2_out[(long)(t * 16 + m) * DD + d] = s2 / (Dg + EPSF);
  }
}

// ---------------------------------------------------------------------------
// Degree-normalized aggregation:
//   out[b] = (sum_k M[row,k] * emb[k]) / (deg + eps)
// idx != 0 : row = idx[blockIdx*16+m]   (gathered rows, transposed==0 path)
// idx == 0 : row = blockIdx*16+m       (full matrix; used with transposed==1,
//            16 consecutive columns per block -> coalesced 64B segments)
// embT is the bf16 transposed embedding [64][8192].
// ---------------------------------------------------------------------------
__global__ __launch_bounds__(256)
void agg_kernel(const int* __restrict__ idx, const float* __restrict__ mat,
                const int transposed, const __bf16* __restrict__ embT,
                float* __restrict__ outp) {
  __shared__ float smem[8192];
  __shared__ float lds_dg[8][16];
  __shared__ int   ridx[16];
  const int lane = threadIdx.x & 31, wv = threadIdx.x >> 5;
  const int half = lane >> 4, ln = lane & 15;
  if (threadIdx.x < 16)
    ridx[threadIdx.x] = idx ? idx[blockIdx.x * 16 + threadIdx.x]
                            : (int)(blockIdx.x * 16 + threadIdx.x);
  __syncthreads();

  const v8f vz = {0.f, 0.f, 0.f, 0.f, 0.f, 0.f, 0.f, 0.f};
  v8f acc[4] = {vz, vz, vz, vz};
  float dgp = 0.f;
  const long myrow = ridx[ln];

  for (int it = 0; it < 32; ++it) {
    const int j0 = wv * 1024 + it * 32;
    v16bf A;
    if (transposed) {
      const long base = (long)j0 * NI + myrow;
      __builtin_prefetch(&mat[base + 32 * (long)NI], 0, 1);
      A = load_a_f32col(mat, base, NI, half, &dgp);
    } else {
      const float* p = mat + myrow * NI + j0 + half * 16;
      __builtin_prefetch(p + 96, 0, 1);
      A = load_a_f32row(p, &dgp);
    }
#pragma unroll
    for (int g = 0; g < 4; ++g) {
      const v16bf B = load_b_bf(embT + (long)(g * 16 + ln) * NI + j0 + 16 * half);
      acc[g] = wmma_bf(A, B, acc[g]);
    }
  }
  dgp += __shfl_xor(dgp, 16, 32);            // combine the two K-halves of row ln
  if (half == 0) lds_dg[wv][ln] = dgp;
#pragma unroll
  for (int g = 0; g < 4; ++g)
#pragma unroll
    for (int r = 0; r < 8; ++r)
      smem[(wv * 16 + r + 8 * half) * 64 + g * 16 + ln] = acc[g][r];
  __syncthreads();
  for (int i2 = (int)threadIdx.x; i2 < 1024; i2 += 256) {
    const int m = i2 >> 6, d = i2 & 63;
    float s = 0.f, Dg = 0.f;
    for (int w = 0; w < 8; ++w) { s += smem[(w * 16 + m) * 64 + d]; Dg += lds_dg[w][m]; }
    outp[(long)(blockIdx.x * 16 + m) * DD + d] = s / (Dg + EPSF);
  }
}

// ---------------------------------------------------------------------------
// Y[n,64] = X[n,64] @ W[64,64]  (optionally tanh).  WT is bf16 W-transposed.
// One wave per 16-row tile; grid*8*16 must equal n.
// ---------------------------------------------------------------------------
__global__ __launch_bounds__(256)
void gemm64_kernel(const float* __restrict__ X, const __bf16* __restrict__ WT,
                   float* __restrict__ Y, const int do_tanh) {
  const int lane = threadIdx.x & 31, half = lane >> 4, ln = lane & 15;
  const int gw = blockIdx.x * 8 + ((int)threadIdx.x >> 5);
  const long m0 = (long)gw * 16;
  const v8f vz = {0.f, 0.f, 0.f, 0.f, 0.f, 0.f, 0.f, 0.f};
  const v16bf A0 = load_a_f32row(X + (m0 + ln) * DD + half * 16, (float*)0);
  const v16bf A1 = load_a_f32row(X + (m0 + ln) * DD + 32 + half * 16, (float*)0);
#pragma unroll
  for (int g = 0; g < 4; ++g) {
    const v16bf B0 = load_b_bf(WT + (long)(g * 16 + ln) * DD + 16 * half);
    const v16bf B1 = load_b_bf(WT + (long)(g * 16 + ln) * DD + 32 + 16 * half);
    v8f c = wmma_bf(A0, B0, vz);
    c = wmma_bf(A1, B1, c);
#pragma unroll
    for (int r = 0; r < 8; ++r) {
      float v = c[r];
      if (do_tanh) v = tanhf(v);
      Y[(m0 + r + 8 * half) * DD + g * 16 + ln] = v;
    }
  }
}

// ---------------------------------------------------------------------------
// Assemble h_user/h_pos/h_neg rows (192 wide) with the item gathers,
// tanh, L2-normalize.  One wave per output row.
// ---------------------------------------------------------------------------
__global__ __launch_bounds__(256)
void finalize_kernel(const float* __restrict__ h1u, const float* __restrict__ h2u,
                     const float* __restrict__ ou,  const float* __restrict__ h2i,
                     const float* __restrict__ oi,
                     const int* __restrict__ pos_it, const int* __restrict__ neg_it,
                     const int* __restrict__ obs_pi, const int* __restrict__ obs_ni,
                     float* __restrict__ out) {
  const int gw = blockIdx.x * 8 + ((int)threadIdx.x >> 5);
  const int lane = threadIdx.x & 31;
  if (gw >= 3 * BB) return;
  const int which = gw >> 10;
  const long b = gw & 1023;
  const float *p0, *p1, *p2;
  if (which == 0) {
    p0 = h1u + b * DD; p1 = h2u + b * DD; p2 = ou + b * DD;
  } else if (which == 1) {
    const long pi = pos_it[b];
    p0 = h2i + pi * DD; p1 = p0; p2 = oi + (long)obs_pi[b] * DD;
  } else {
    const long ni = neg_it[b];
    p0 = h2i + ni * DD; p1 = p0; p2 = oi + (long)obs_ni[b] * DD;
  }
  float v[6];
  float ss = 0.f;
#pragma unroll
  for (int e = 0; e < 6; ++e) {
    const int c = e * 32 + lane;             // 0..191
    const float x0 = (c < 64) ? p0[c] : (c < 128 ? p1[c - 64] : p2[c - 128]);
    const float x = tanhf(x0);
    v[e] = x; ss += x * x;
  }
#pragma unroll
  for (int off = 1; off < 32; off <<= 1) ss += __shfl_xor(ss, off, 32);
  const float inv = 1.0f / fmaxf(sqrtf(ss), 1e-12f);
#pragma unroll
  for (int e = 0; e < 6; ++e) out[(long)gw * 192 + e * 32 + lane] = v[e] * inv;
}

// ---------------------------------------------------------------------------
extern "C" void kernel_launch(void* const* d_in, const int* in_sizes, int n_in,
                              void* d_out, int out_size, void* d_ws, size_t ws_size,
                              hipStream_t stream) {
  (void)in_sizes; (void)n_in; (void)out_size; (void)ws_size;
  const int*   users    = (const int*)d_in[0];
  const int*   pos_it   = (const int*)d_in[1];
  const int*   neg_it   = (const int*)d_in[2];
  const float* adj      = (const float*)d_in[3];
  const int*   obs_u    = (const int*)d_in[4];
  const int*   obs_pi   = (const int*)d_in[5];
  const int*   obs_ni   = (const int*)d_in[6];
  const float* obs_adj  = (const float*)d_in[7];
  const float* user_emb = (const float*)d_in[9];
  const float* item_emb = (const float*)d_in[10];
  const float* W1       = (const float*)d_in[11];
  const float* W2       = (const float*)d_in[12];
  const float* Wobs     = (const float*)d_in[13];

  float* ws = (float*)d_ws;
  const long GSZ = (long)BB * DD;            // 65536 floats (gathered buffers)
  const long FSZ = (long)NI * DD;            // 524288 floats (full buffers)
  float* a1   = ws + 0 * GSZ;                // coef-agg (users)
  float* a2   = ws + 1 * GSZ;                // adj-agg (users)
  float* cu   = ws + 2 * GSZ;                // obs_adj-agg (obs_users)
  float* h1u  = ws + 3 * GSZ;
  float* h2u  = ws + 4 * GSZ;
  float* ou   = ws + 5 * GSZ;
  float* bi   = ws + 6 * GSZ;                // full adj^T agg      [8192,64]
  float* ci   = ws + 6 * GSZ + 1 * FSZ;      // full obs_adj^T agg  [8192,64]
  float* h2i  = ws + 6 * GSZ + 2 * FSZ;      // h2_item full        [8192,64]
  float* oi   = ws + 6 * GSZ + 3 * FSZ;      // obs_item full       [8192,64]

  // bf16 region (16B-aligned)
  __bf16* bfb   = (__bf16*)(ws + 6 * GSZ + 4 * FSZ);
  const long EM = (long)NU * DD;             // 524288 elements
  __bf16* ue_bf = bfb + 0 * EM;              // user_emb bf16 row-major
  __bf16* ie_bf = bfb + 1 * EM;              // item_emb bf16 row-major
  __bf16* ueT   = bfb + 2 * EM;              // user_emb bf16 transposed [64][8192]
  __bf16* ieT   = bfb + 3 * EM;              // item_emb bf16 transposed [64][8192]
  __bf16* W1T   = bfb + 4 * EM;
  __bf16* W2T   = W1T + DD * DD;
  __bf16* WoT   = W2T + DD * DD;

  // --- prep: one-time bf16 conversions ---
  cvt_emb_kernel<<<2048, 256, 0, stream>>>(user_emb, ue_bf, ueT, NU);
  cvt_emb_kernel<<<2048, 256, 0, stream>>>(item_emb, ie_bf, ieT, NI);
  cvt_wT_kernel<<<16, 256, 0, stream>>>(W1, W1T);
  cvt_wT_kernel<<<16, 256, 0, stream>>>(W2, W2T);
  cvt_wT_kernel<<<16, 256, 0, stream>>>(Wobs, WoT);

  // --- aggregation passes ---
  // gathered user-row passes (32MB coalesced adjacency reads each)
  attn_user_kernel<<<64, 256, 0, stream>>>(users, ue_bf, ie_bf, ieT, adj, a1, a2);
  agg_kernel<<<64, 256, 0, stream>>>(obs_u, obs_adj, 0, ieT, cu);
  // full column passes: each matrix streamed exactly once, coalesced;
  // one pass serves both pos and neg gathers
  agg_kernel<<<512, 256, 0, stream>>>((const int*)0, adj, 1, ueT, bi);
  agg_kernel<<<512, 256, 0, stream>>>((const int*)0, obs_adj, 1, ueT, ci);

  // --- W GEMMs ---
  gemm64_kernel<<<8, 256, 0, stream>>>(a1, W1T, h1u, 0);
  gemm64_kernel<<<8, 256, 0, stream>>>(a2, W2T, h2u, 0);
  gemm64_kernel<<<8, 256, 0, stream>>>(cu, WoT, ou, 1);
  gemm64_kernel<<<64, 256, 0, stream>>>(bi, W2T, h2i, 0);
  gemm64_kernel<<<64, 256, 0, stream>>>(ci, WoT, oi, 1);

  finalize_kernel<<<384, 256, 0, stream>>>(h1u, h2u, ou, h2i, oi,
                                           pos_it, neg_it, obs_pi, obs_ni,
                                           (float*)d_out);
}
